// RSSM_23210003267912
// MI455X (gfx1250) — matrix operand order repaired and checked by reading, not compiled, not run on hardware
//
#include <hip/hip_runtime.h>

typedef __attribute__((ext_vector_type(16))) __bf16 v16bf;
typedef __attribute__((ext_vector_type(8)))  float  v8f;

union FragBF16 {
    uint4 u[2];
    v16bf v;
};

__device__ __forceinline__ unsigned short f32_to_bf16(float f) {
    unsigned int u = __float_as_uint(f);
    u += 0x7FFFu + ((u >> 16) & 1u);   // round-to-nearest-even
    return (unsigned short)(u >> 16);
}
__device__ __forceinline__ float bf16_to_f32(unsigned short h) {
    return __uint_as_float(((unsigned int)h) << 16);
}
__device__ __forceinline__ float elu_f(float v) {
    return v > 0.f ? v : (__expf(v) - 1.f);
}
__device__ __forceinline__ float softplus_f(float v) {
    return v > 20.f ? v : log1pf(__expf(v));
}
__device__ __forceinline__ float sigmoid_f(float v) {
    return 1.f / (1.f + __expf(-v));
}

// ---------------------------------------------------------------------------
// Generic bf16 WMMA GEMM: C[M x N] = act(A[M x K] * W[N x K]^T + bias)
// A row-major (K contig), W row-major (K contig) == B column-major.
// MODE 0: f32 output, no activation. MODE 1: ELU -> bf16 output.
// Block tile 128(M) x 64(N), 256 threads = 8 waves, each wave 32x32 (2x2 WMMA).
// Double-buffered LDS: exactly one barrier per K-chunk (no WAR barrier needed:
// iteration kc reads buf[kc&1] while its tail stores buf[kc&1 ^ 1]).
// MODE is a template parameter so the epilogue is straight-line code.
// ---------------------------------------------------------------------------
#define LDS_STRIDE 40   // bf16 elems per row (padded from 32 to spread banks)

template <int MODE>
__global__ __launch_bounds__(256) void gemm_bf16_wmma(
    const unsigned short* __restrict__ A,
    const unsigned short* __restrict__ W,
    const float* __restrict__ bias,
    void* __restrict__ Cout,
    int N, int K)
{
    __shared__ unsigned short As[2][128 * LDS_STRIDE];
    __shared__ unsigned short Bs[2][64 * LDS_STRIDE];

    const int tid  = threadIdx.x;
    const int lane = tid & 31;
    const int wave = tid >> 5;
    const int wrow = wave & 3;   // wave's 32-row slot (0..3)
    const int wcol = wave >> 2;  // wave's 32-col slot (0..1)
    const int bm   = blockIdx.y * 128;
    const int bn   = blockIdx.x * 64;

    // Per-thread staging slots (fixed across iterations)
    const int a0row = tid >> 2;
    const int a0kk  = (tid & 3) << 3;
    const int a1row = (tid + 256) >> 2;
    const int brow  = tid >> 2;
    const int bkk   = (tid & 3) << 3;

    v8f acc[2][2] = {};

    const int nk = K >> 5;   // K / 32 (all K are multiples of 32)

    // ---- prologue: fetch chunk 0 into regs, store to buffer 0 ----
    uint4 ra0 = *(const uint4*)(A + (size_t)(bm + a0row) * K + a0kk);
    uint4 ra1 = *(const uint4*)(A + (size_t)(bm + a1row) * K + a0kk);
    uint4 rb  = *(const uint4*)(W + (size_t)(bn + brow)  * K + bkk);
    *(uint4*)(&As[0][a0row * LDS_STRIDE + a0kk]) = ra0;
    *(uint4*)(&As[0][a1row * LDS_STRIDE + a0kk]) = ra1;
    *(uint4*)(&Bs[0][brow  * LDS_STRIDE + bkk])  = rb;

    for (int kc = 0; kc < nk; ++kc) {
        const int buf = kc & 1;
        const bool more = (kc + 1) < nk;
        // issue global loads for next chunk early (overlap with barrier+WMMA)
        if (more) {
            int k0 = (kc + 1) << 5;
            ra0 = *(const uint4*)(A + (size_t)(bm + a0row) * K + k0 + a0kk);
            ra1 = *(const uint4*)(A + (size_t)(bm + a1row) * K + k0 + a0kk);
            rb  = *(const uint4*)(W + (size_t)(bn + brow)  * K + k0 + bkk);
        }
        __syncthreads();   // stores to buf (from prologue / previous tail) visible

        // --- build fragments per CDNA5 16-bit layouts ---
        FragBF16 afrag[2], bfrag[2];
#pragma unroll
        for (int i = 0; i < 2; ++i) {
            int m  = wrow * 32 + i * 16 + (lane & 15);
            int kb = (lane >> 4) * 8;   // lanes16-31 hold K 8..15 / 24..31
            afrag[i].u[0] = *(const uint4*)(&As[buf][m * LDS_STRIDE + kb]);
            afrag[i].u[1] = *(const uint4*)(&As[buf][m * LDS_STRIDE + 16 + kb]);
        }
#pragma unroll
        for (int j = 0; j < 2; ++j) {
            int n  = wcol * 32 + j * 16 + (lane & 15);
            int kb = (lane >> 4) * 8;   // lanes16-31 hold K 8..15 / 24..31
            bfrag[j].u[0] = *(const uint4*)(&Bs[buf][n * LDS_STRIDE + kb]);
            bfrag[j].u[1] = *(const uint4*)(&Bs[buf][n * LDS_STRIDE + 16 + kb]);
        }

#pragma unroll
        for (int i = 0; i < 2; ++i)
#pragma unroll
            for (int j = 0; j < 2; ++j)
                acc[i][j] = __builtin_amdgcn_wmma_f32_16x16x32_bf16(
                    false, afrag[i].v, false, bfrag[j].v,
                    (short)0, acc[i][j], false, false);

        // tail: store next chunk into the other buffer (no barrier needed:
        // same-wave LDS ops are in-order; other waves only touch buf)
        if (more) {
            const int nb = buf ^ 1;
            *(uint4*)(&As[nb][a0row * LDS_STRIDE + a0kk]) = ra0;
            *(uint4*)(&As[nb][a1row * LDS_STRIDE + a0kk]) = ra1;
            *(uint4*)(&Bs[nb][brow  * LDS_STRIDE + bkk])  = rb;
        }
    }

    // --- epilogue: C/D layout lane<16 -> M=e, lane>=16 -> M=e+8; N = lane&15 ---
#pragma unroll
    for (int i = 0; i < 2; ++i) {
#pragma unroll
        for (int j = 0; j < 2; ++j) {
            int mbase = bm + wrow * 32 + i * 16 + 8 * (lane >> 4);
            int nidx  = bn + wcol * 32 + j * 16 + (lane & 15);
            float bv  = bias[nidx];
#pragma unroll
            for (int e = 0; e < 8; ++e) {
                float v = acc[i][j][e] + bv;
                size_t off = (size_t)(mbase + e) * N + nidx;
                if (MODE == 1) {
                    ((unsigned short*)Cout)[off] = f32_to_bf16(elu_f(v));
                } else {
                    ((float*)Cout)[off] = v;
                }
            }
        }
    }
}

// ---------------------------------------------------------------------------
// Elementwise kernels
// ---------------------------------------------------------------------------
__global__ void convert_f32_bf16(const float* __restrict__ src,
                                 unsigned short* __restrict__ dst, int n) {
    int i = blockIdx.x * 256 + threadIdx.x;
    if (i < n) dst[i] = f32_to_bf16(src[i]);
}

// build xa = bf16([state*nt, a]) [512x160], pb = belief*nt (f32 + bf16) [512x512]
// 2-D grid: blockIdx.y = row (batch), x covers 672 columns (3 blocks of 256).
__global__ void prep_input(const float* __restrict__ state,
                           const float* __restrict__ belief,
                           const float* __restrict__ act,
                           const float* __restrict__ nt,
                           unsigned short* __restrict__ xa,
                           unsigned short* __restrict__ pb_bf,
                           float* __restrict__ pb_f) {
    int col = blockIdx.x * 256 + threadIdx.x;   // 0..767 (guard at 672)
    int row = blockIdx.y;
    if (col >= 672) return;
    float ntv = nt[row];
    if (col < 128) {
        xa[row * 160 + col] = f32_to_bf16(state[row * 128 + col] * ntv);
    } else if (col < 160) {
        xa[row * 160 + col] = f32_to_bf16(act[row * 32 + (col - 128)]);
    } else {
        int c = col - 160;
        float v = belief[row * 512 + c] * ntv;
        pb_f[row * 512 + c] = v;
        pb_bf[row * 512 + c] = f32_to_bf16(v);
    }
}

// GRU gates: b1 = (1-z)*n + z*h ; write beliefs f32, b1 bf16, and r_in[:,128:640]
__global__ void gru_gates(const float* __restrict__ gi,
                          const float* __restrict__ gh,
                          const float* __restrict__ pb,
                          float* __restrict__ beliefs,
                          unsigned short* __restrict__ b1_bf,
                          unsigned short* __restrict__ rin) {
    int i = blockIdx.x * 256 + threadIdx.x;
    if (i >= 512 * 512) return;
    int row = i >> 9, c = i & 511;
    const float* gir = gi + (size_t)row * 1536;
    const float* ghr = gh + (size_t)row * 1536;
    float r = sigmoid_f(gir[c] + ghr[c]);
    float z = sigmoid_f(gir[512 + c] + ghr[512 + c]);
    float n = tanhf(gir[1024 + c] + r * ghr[1024 + c]);
    float b1 = (1.f - z) * n + z * pb[i];
    beliefs[i] = b1;
    unsigned short bb = f32_to_bf16(b1);
    b1_bf[i] = bb;
    rin[(size_t)row * 640 + 128 + c] = bb;
}

// mean/std/state from [512 x 256] (mean | log-std) block; optionally feed r_in
template <bool FEED_RIN>
__global__ void dist_stats(const float* __restrict__ mo,
                           const float* __restrict__ noise,
                           float* __restrict__ mean_o,
                           float* __restrict__ std_o,
                           float* __restrict__ state_o,
                           unsigned short* __restrict__ rin) {
    int i = blockIdx.x * 256 + threadIdx.x;
    if (i >= 512 * 128) return;
    int row = i >> 7, c = i & 127;
    float m   = mo[(size_t)row * 256 + c];
    float ls  = mo[(size_t)row * 256 + 128 + c];
    float sd  = softplus_f(ls) + 0.1f;
    float st  = m + noise[i] * sd;
    mean_o[i]  = m;
    std_o[i]   = sd;
    state_o[i] = st;
    if (FEED_RIN) rin[(size_t)row * 640 + c] = f32_to_bf16(st);
}

// posterior input: [o(1024) | b1(512)] -> bf16 [512 x 1536]
// 2-D grid: blockIdx.y = row, x covers 1536 columns (6 blocks of 256).
__global__ void post_input(const float* __restrict__ obs,
                           const unsigned short* __restrict__ b1_bf,
                           unsigned short* __restrict__ po) {
    int col = blockIdx.x * 256 + threadIdx.x;   // 0..1535
    int row = blockIdx.y;
    po[(size_t)row * 1536 + col] =
        (col < 1024) ? f32_to_bf16(obs[(size_t)row * 1024 + col])
                     : b1_bf[(size_t)row * 512 + (col - 1024)];
}

// final reward: rew[row] = dot(rh2[row,:512], Wr3) + br3  (one wave per row)
__global__ __launch_bounds__(256) void reward_final(
    const unsigned short* __restrict__ rh,
    const unsigned short* __restrict__ w3,
    const float* __restrict__ br3,
    float* __restrict__ out) {
    int wave = threadIdx.x >> 5, lane = threadIdx.x & 31;
    int row = blockIdx.x * 8 + wave;
    float sum = 0.f;
#pragma unroll
    for (int j = 0; j < 16; ++j) {
        int k = lane + 32 * j;
        sum += bf16_to_f32(rh[(size_t)row * 512 + k]) * bf16_to_f32(w3[k]);
    }
#pragma unroll
    for (int off = 16; off > 0; off >>= 1)
        sum += __shfl_down(sum, off, 32);
    if (lane == 0) out[row] = sum + br3[0];
}

// ---------------------------------------------------------------------------
// Host launcher
// ---------------------------------------------------------------------------
extern "C" void kernel_launch(void* const* d_in, const int* in_sizes, int n_in,
                              void* d_out, int out_size, void* d_ws, size_t ws_size,
                              hipStream_t stream) {
    (void)in_sizes; (void)n_in; (void)out_size; (void)ws_size;
    const int SEQ = 64, BATCH = 512, S = 128, A = 32, B = 512, E = 1024, H = 512;

    const float* prev_state   = (const float*)d_in[0];
    const float* actions      = (const float*)d_in[1];
    const float* prev_beliefs = (const float*)d_in[2];
    const float* observations = (const float*)d_in[3];
    const float* non_term     = (const float*)d_in[4];
    const float* prior_noise  = (const float*)d_in[5];
    const float* post_noise   = (const float*)d_in[6];
    const float* We = (const float*)d_in[7];  const float* be = (const float*)d_in[8];
    const float* Wi = (const float*)d_in[9];  const float* bi = (const float*)d_in[10];
    const float* Wh = (const float*)d_in[11]; const float* bh = (const float*)d_in[12];
    const float* Wt1 = (const float*)d_in[13]; const float* bt1 = (const float*)d_in[14];
    const float* Wt2 = (const float*)d_in[15]; const float* bt2 = (const float*)d_in[16];
    const float* Wp1 = (const float*)d_in[17]; const float* bp1 = (const float*)d_in[18];
    const float* Wp2 = (const float*)d_in[19]; const float* bp2 = (const float*)d_in[20];
    const float* Wr1 = (const float*)d_in[21]; const float* br1 = (const float*)d_in[22];
    const float* Wr2 = (const float*)d_in[23]; const float* br2 = (const float*)d_in[24];
    const float* Wr3 = (const float*)d_in[25]; const float* br3 = (const float*)d_in[26];

    // workspace carve-out (256B aligned)
    char* ws = (char*)d_ws;
    size_t off = 0;
    auto carve = [&](size_t bytes) -> char* {
        char* p = ws + off;
        off = (off + bytes + 255) & ~(size_t)255;
        return p;
    };
    unsigned short* We_bf  = (unsigned short*)carve((size_t)B * (S + A) * 2);
    unsigned short* Wi_bf  = (unsigned short*)carve((size_t)3 * B * B * 2);
    unsigned short* Wh_bf  = (unsigned short*)carve((size_t)3 * B * B * 2);
    unsigned short* Wt1_bf = (unsigned short*)carve((size_t)H * B * 2);
    unsigned short* Wt2_bf = (unsigned short*)carve((size_t)2 * S * H * 2);
    unsigned short* Wp1_bf = (unsigned short*)carve((size_t)H * (B + E) * 2);
    unsigned short* Wp2_bf = (unsigned short*)carve((size_t)2 * S * H * 2);
    unsigned short* Wr1_bf = (unsigned short*)carve((size_t)H * (S + B) * 2);
    unsigned short* Wr2_bf = (unsigned short*)carve((size_t)H * H * 2);
    unsigned short* Wr3_bf = (unsigned short*)carve((size_t)H * 2);

    unsigned short* xa_bf    = (unsigned short*)carve((size_t)BATCH * 160 * 2);
    unsigned short* pb_bf    = (unsigned short*)carve((size_t)BATCH * B * 2);
    float*          pb_f     = (float*)carve((size_t)BATCH * B * 4);
    unsigned short* x_bf     = (unsigned short*)carve((size_t)BATCH * B * 2);
    float*          gi_f     = (float*)carve((size_t)BATCH * 3 * B * 4);
    float*          gh_f     = (float*)carve((size_t)BATCH * 3 * B * 4);
    unsigned short* b1_bf    = (unsigned short*)carve((size_t)BATCH * B * 2);
    unsigned short* th_bf    = (unsigned short*)carve((size_t)BATCH * H * 2);
    float*          t_out    = (float*)carve((size_t)BATCH * 2 * S * 4);
    unsigned short* po_in_bf = (unsigned short*)carve((size_t)BATCH * (B + E) * 2);
    unsigned short* ph_bf    = (unsigned short*)carve((size_t)BATCH * H * 2);
    float*          q_out    = (float*)carve((size_t)BATCH * 2 * S * 4);
    unsigned short* r_in_bf  = (unsigned short*)carve((size_t)BATCH * (S + B) * 2);
    unsigned short* rh1_bf   = (unsigned short*)carve((size_t)BATCH * H * 2);
    unsigned short* rh2_bf   = (unsigned short*)carve((size_t)BATCH * H * 2);

    // output offsets (floats), tuple order
    float* out = (float*)d_out;
    const size_t O_BEL  = 0;
    const size_t O_PS   = (size_t)SEQ * BATCH * B;
    const size_t O_PM   = O_PS   + (size_t)SEQ * BATCH * S;
    const size_t O_PSTD = O_PM   + (size_t)SEQ * BATCH * S;
    const size_t O_QS   = O_PSTD + (size_t)SEQ * BATCH * S;
    const size_t O_QM   = O_QS   + (size_t)SEQ * BATCH * S;
    const size_t O_QSTD = O_QM   + (size_t)SEQ * BATCH * S;
    const size_t O_R    = O_QSTD + (size_t)SEQ * BATCH * S;

    // weight conversion (f32 -> bf16), once per launch
    auto conv = [&](const float* src, unsigned short* dst, int n) {
        convert_f32_bf16<<<(n + 255) / 256, 256, 0, stream>>>(src, dst, n);
    };
    conv(We,  We_bf,  B * (S + A));
    conv(Wi,  Wi_bf,  3 * B * B);
    conv(Wh,  Wh_bf,  3 * B * B);
    conv(Wt1, Wt1_bf, H * B);
    conv(Wt2, Wt2_bf, 2 * S * H);
    conv(Wp1, Wp1_bf, H * (B + E));
    conv(Wp2, Wp2_bf, 2 * S * H);
    conv(Wr1, Wr1_bf, H * (S + B));
    conv(Wr2, Wr2_bf, H * H);
    conv(Wr3, Wr3_bf, H);

    auto gemm_f32 = [&](const unsigned short* Am, const unsigned short* Wm,
                        const float* bv, float* Cm, int N, int K) {
        dim3 grid(N / 64, BATCH / 128);
        gemm_bf16_wmma<0><<<grid, 256, 0, stream>>>(Am, Wm, bv, Cm, N, K);
    };
    auto gemm_elu = [&](const unsigned short* Am, const unsigned short* Wm,
                        const float* bv, unsigned short* Cm, int N, int K) {
        dim3 grid(N / 64, BATCH / 128);
        gemm_bf16_wmma<1><<<grid, 256, 0, stream>>>(Am, Wm, bv, Cm, N, K);
    };

    for (int t = 0; t < SEQ; ++t) {
        const float* state_prev  = (t == 0) ? prev_state
                                            : out + O_QS  + (size_t)(t - 1) * BATCH * S;
        const float* belief_prev = (t == 0) ? prev_beliefs
                                            : out + O_BEL + (size_t)(t - 1) * BATCH * B;
        const float* a_t  = actions      + (size_t)t * BATCH * A;
        const float* o_t  = observations + (size_t)t * BATCH * E;
        const float* nt_t = non_term     + (size_t)t * BATCH;
        const float* pn_t = prior_noise  + (size_t)t * BATCH * S;
        const float* qn_t = post_noise   + (size_t)t * BATCH * S;

        prep_input<<<dim3(3, BATCH), 256, 0, stream>>>(
            state_prev, belief_prev, a_t, nt_t, xa_bf, pb_bf, pb_f);

        // x = elu(xa @ We^T + be)
        gemm_elu(xa_bf, We_bf, be, x_bf, B, S + A);
        // GRU input/hidden projections
        gemm_f32(x_bf,  Wi_bf, bi, gi_f, 3 * B, B);
        gemm_f32(pb_bf, Wh_bf, bh, gh_f, 3 * B, B);
        gru_gates<<<(BATCH * B) / 256, 256, 0, stream>>>(
            gi_f, gh_f, pb_f, out + O_BEL + (size_t)t * BATCH * B, b1_bf, r_in_bf);

        // transition
        gemm_elu(b1_bf, Wt1_bf, bt1, th_bf, H, B);
        gemm_f32(th_bf, Wt2_bf, bt2, t_out, 2 * S, H);
        dist_stats<false><<<(BATCH * S) / 256, 256, 0, stream>>>(
            t_out, pn_t,
            out + O_PM   + (size_t)t * BATCH * S,
            out + O_PSTD + (size_t)t * BATCH * S,
            out + O_PS   + (size_t)t * BATCH * S,
            nullptr);

        // posterior
        post_input<<<dim3(6, BATCH), 256, 0, stream>>>(o_t, b1_bf, po_in_bf);
        gemm_elu(po_in_bf, Wp1_bf, bp1, ph_bf, H, B + E);
        gemm_f32(ph_bf,    Wp2_bf, bp2, q_out, 2 * S, H);
        dist_stats<true><<<(BATCH * S) / 256, 256, 0, stream>>>(
            q_out, qn_t,
            out + O_QM   + (size_t)t * BATCH * S,
            out + O_QSTD + (size_t)t * BATCH * S,
            out + O_QS   + (size_t)t * BATCH * S,
            r_in_bf);

        // reward MLP
        gemm_elu(r_in_bf, Wr1_bf, br1, rh1_bf, H, S + B);
        gemm_elu(rh1_bf,  Wr2_bf, br2, rh2_bf, H, H);
        reward_final<<<BATCH / 8, 256, 0, stream>>>(
            rh2_bf, Wr3_bf, br3, out + O_R + (size_t)t * BATCH);
    }
}
